// RGCN_9895604650659
// MI455X (gfx1250) — matrix operand order repaired and checked by reading, not compiled, not run on hardware
//
#include <hip/hip_runtime.h>
#include <hip/hip_bf16.h>
#include <stdint.h>

typedef __bf16 bf16_t;
typedef __attribute__((ext_vector_type(16))) __bf16 v16bf;
typedef __attribute__((ext_vector_type(8)))  __bf16 v8bf;
typedef __attribute__((ext_vector_type(8)))  float   v8f;

#define NN    50000
#define NREL  4
#define NE    250000
#define INF   512
#define HIDF  512
#define OUTF  256
#define KDIM  512      // K = 512 for both layers
#define F2    256      // packed bf16 pairs per node row (512 feats / 2)

// ---------------------------------------------------------------------------
// Elementwise / prep kernels
// ---------------------------------------------------------------------------

__global__ void zero_u32_kernel(uint32_t* __restrict__ p, long long n) {
  long long i = (long long)blockIdx.x * blockDim.x + threadIdx.x;
  if (i < n) p[i] = 0u;
}

__global__ void f32_to_bf16_kernel(const float* __restrict__ in, bf16_t* __restrict__ out, long long n) {
  long long i = (long long)blockIdx.x * blockDim.x + threadIdx.x;
  if (i < n) out[i] = (bf16_t)in[i];
}

__global__ void relu_f32_to_bf16_kernel(const float* __restrict__ in, bf16_t* __restrict__ out, long long n) {
  long long i = (long long)blockIdx.x * blockDim.x + threadIdx.x;
  if (i < n) out[i] = (bf16_t)fmaxf(in[i], 0.0f);
}

// Wt[n*K + k] = bf16( sum_r W[r][k][n] )   (summed self weights, transposed)
__global__ void sum_self_transpose_kernel(const float* __restrict__ W, bf16_t* __restrict__ Wt,
                                          int K, int N) {
  long long i = (long long)blockIdx.x * blockDim.x + threadIdx.x;
  if (i >= (long long)K * N) return;
  int n = (int)(i % N);
  int k = (int)(i / N);
  float s = 0.0f;
  for (int r = 0; r < NREL; ++r) s += W[((long long)r * K + k) * N + n];
  Wt[(long long)n * K + k] = (bf16_t)s;
}

// Wt[r][n*K + k] = bf16( W[r][k][n] )
__global__ void transpose_rel_kernel(const float* __restrict__ W, bf16_t* __restrict__ Wt,
                                     int K, int N) {
  long long i = (long long)blockIdx.x * blockDim.x + threadIdx.x;
  if (i >= (long long)NREL * K * N) return;
  int n = (int)(i % N);
  long long t = i / N;
  int k = (int)(t % K);
  int r = (int)(t / K);
  Wt[((long long)r * N + n) * K + k] = (bf16_t)W[((long long)r * K + k) * N + n];
}

__global__ void bias_sum_kernel(const float* __restrict__ b, float* __restrict__ bs, int N) {
  int n = blockIdx.x * blockDim.x + threadIdx.x;
  if (n >= N) return;
  float s = 0.0f;
  for (int r = 0; r < NREL; ++r) s += b[r * N + n];
  bs[n] = s;
}

// C[m][n] = bs[n] for all m
__global__ void init_c_bias_kernel(float* __restrict__ C, const float* __restrict__ bs,
                                   long long total, int N) {
  long long i = (long long)blockIdx.x * blockDim.x + threadIdx.x;
  if (i < total) C[i] = bs[(int)(i % N)];
}

// deg[r][dst] += 1 for all relations at once (edge arrays are [NREL][NE] flat)
__global__ void deg_kernel(const long long* __restrict__ dst, float* __restrict__ deg) {
  long long i = (long long)blockIdx.x * blockDim.x + threadIdx.x;
  if (i >= (long long)NREL * NE) return;
  int r = (int)(i / NE);
  int d = (int)dst[i];
  atomicAdd(&deg[(long long)r * NN + d], 1.0f);
}

// ---------------------------------------------------------------------------
// Neighbor aggregation: packed-bf16 scatter-add via GLOBAL_ATOMIC_PK_ADD_BF16
// One block per edge -> src/dst indices become uniform scalar loads.
// ---------------------------------------------------------------------------

__global__ __launch_bounds__(256) void scatter_pk_bf16_kernel(
    const uint32_t* __restrict__ h2,   // [NN][F2] packed bf16x2
    const long long* __restrict__ src,
    const long long* __restrict__ dst,
    uint32_t* __restrict__ acc2) {     // [NN][F2]
  const int e = blockIdx.x;            // uniform -> s_load for src/dst
  const int f = threadIdx.x;           // 0..255
  const int s = (int)src[e];
  const int d = (int)dst[e];
  uint32_t v = h2[(long long)s * F2 + f];
  unsigned long long addr = (unsigned long long)(uintptr_t)&acc2[(long long)d * F2 + f];
  // CDNA5 packed bf16 atomic add (no-return -> STOREcnt, waited at s_endpgm)
  asm volatile("global_atomic_pk_add_bf16 %0, %1, off" :: "v"(addr), "v"(v) : "memory");
}

// in-place: acc2 = acc2 / max(deg,1). One block per node -> deg is scalar load.
__global__ __launch_bounds__(256) void mean_div_kernel(
    uint32_t* __restrict__ acc2, const float* __restrict__ deg) {
  const int node = blockIdx.x;
  const int f = threadIdx.x;
  float inv = 1.0f / fmaxf(deg[node], 1.0f);
  long long i = (long long)node * F2 + f;
  uint32_t v = acc2[i];
  float lo = __uint_as_float(v << 16) * inv;
  float hi = __uint_as_float(v & 0xffff0000u) * inv;
  unsigned short slo = __builtin_bit_cast(unsigned short, (bf16_t)lo);
  unsigned short shi = __builtin_bit_cast(unsigned short, (bf16_t)hi);
  acc2[i] = (uint32_t)slo | ((uint32_t)shi << 16);
}

// ---------------------------------------------------------------------------
// WMMA bf16 GEMM:  C[M][N] += A[M][K=512] * Wt[N][K=512]^T   (fp32 accumulate)
// Block: 256 threads (8 waves), tile 128x128, K-step 32, DOUBLE-BUFFERED LDS.
// Wave grid 2(M) x 4(N); each wave: 64x32 = 4x2 fragments of 16x16.
// Tile staging via GLOBAL_LOAD_ASYNC_TO_LDS_B128 (ASYNCcnt): async loads for
// k-step i+1 are issued before computing step i, latency hidden behind WMMAs;
// single s_wait_asynccnt + barrier per step covers both buffer hand-offs.
// ---------------------------------------------------------------------------

#define MT    128
#define NT    128
#define KT    32
#define LDT   40   // padded LDS row stride (elements) -> conflict-free frag reads
#define NSTEP (KDIM / KT)

union V16U { v16bf v; v8bf h[2]; };

__device__ __forceinline__ void async_tile_b128(void* lds_dst, const void* gsrc) {
  // LDS generic pointers carry the LDS byte offset in their low 32 bits
  // (ISA: LDS aperture address truncates to addr[31:0]).
  uint32_t lds_off = (uint32_t)(uintptr_t)lds_dst;
  unsigned long long gaddr = (unsigned long long)(uintptr_t)gsrc;
  asm volatile("global_load_async_to_lds_b128 %0, %1, off"
               :: "v"(lds_off), "v"(gaddr) : "memory");
}

__device__ __forceinline__ void wait_async_zero() {
  asm volatile("s_wait_asynccnt 0" ::: "memory");
}

__device__ __forceinline__ void issue_tile_pair(bf16_t* asw, bf16_t* bsw,
                                                const bf16_t* ap, const bf16_t* bp) {
  async_tile_b128(asw,     ap);
  async_tile_b128(asw + 8, ap + 8);
  async_tile_b128(bsw,     bp);
  async_tile_b128(bsw + 8, bp + 8);
}

__global__ __launch_bounds__(256) void wmma_gemm_acc_kernel(
    const bf16_t* __restrict__ A,    // [M][KDIM] row-major bf16
    const bf16_t* __restrict__ Wt,   // [N][KDIM] row-major bf16 (pre-transposed weights)
    float* __restrict__ C,           // [M][N] fp32, accumulated in-place
    int M, int N) {
  __shared__ __align__(16) bf16_t As[2][MT * LDT];
  __shared__ __align__(16) bf16_t Bs[2][NT * LDT];

  const int tid  = threadIdx.x;
  const int wid  = tid >> 5;
  const int lane = tid & 31;
  const int wm   = wid & 1;          // 0..1  (64 rows each)
  const int wn   = wid >> 1;         // 0..3  (32 cols each)
  const int blockM = blockIdx.y * MT;
  const int blockN = blockIdx.x * NT;

  v8f acc[4][2] = {};

  // cooperative tile loads: 256 threads, 2 threads per row, 16 bf16 (32B) each
  const int lrow = tid >> 1;                 // 0..127
  const int lcol = (tid & 1) * 16;           // 0 or 16
  int arow = blockM + lrow; if (arow >= M) arow = M - 1;   // clamp (store predicated later)
  const bf16_t* Aptr = A  + (size_t)arow * KDIM + lcol;
  const bf16_t* Bptr = Wt + (size_t)(blockN + lrow) * KDIM + lcol;
  bf16_t* AsW[2] = { &As[0][lrow * LDT + lcol], &As[1][lrow * LDT + lcol] };
  bf16_t* BsW[2] = { &Bs[0][lrow * LDT + lcol], &Bs[1][lrow * LDT + lcol] };

  // A frag coords (ISA 16-bit A 16x32: lane<16 -> K{0..7,16..23}; lane>=16 -> K{8..15,24..31})
  const int mrow   = wm * 64 + (lane & 15);
  const int khalfA = (lane >> 4) * 8;
  // B frag coords (same K order as A: lane<16 -> K{0..7,16..23}; lane>=16 -> K{8..15,24..31})
  const int nrow   = wn * 32 + (lane & 15);
  const int kbaseB = (lane >> 4) * 8;

  // prologue: stage k-step 0 into buffer 0
  issue_tile_pair(AsW[0], BsW[0], Aptr, Bptr);
  wait_async_zero();
  __syncthreads();

#pragma unroll
  for (int step = 0; step < NSTEP; ++step) {
    const int buf = step & 1;
    // prefetch next k-step into the other buffer (no wait -> overlaps WMMAs)
    if (step + 1 < NSTEP)
      issue_tile_pair(AsW[buf ^ 1], BsW[buf ^ 1],
                      Aptr + (step + 1) * KT, Bptr + (step + 1) * KT);

    V16U afrag[4], bfrag[2];
#pragma unroll
    for (int fm = 0; fm < 4; ++fm) {
      const bf16_t* p = &As[buf][(mrow + fm * 16) * LDT + khalfA];
      afrag[fm].h[0] = *(const v8bf*)p;          // K = khalfA .. khalfA+7
      afrag[fm].h[1] = *(const v8bf*)(p + 16);   // K = khalfA+16 .. khalfA+23
    }
#pragma unroll
    for (int fn = 0; fn < 2; ++fn) {
      const bf16_t* p = &Bs[buf][(nrow + fn * 16) * LDT + kbaseB];
      bfrag[fn].h[0] = *(const v8bf*)p;          // K = kbaseB .. kbaseB+7
      bfrag[fn].h[1] = *(const v8bf*)(p + 16);   // K = kbaseB+16 .. kbaseB+23
    }
#pragma unroll
    for (int fm = 0; fm < 4; ++fm)
#pragma unroll
      for (int fn = 0; fn < 2; ++fn)
        acc[fm][fn] = __builtin_amdgcn_wmma_f32_16x16x32_bf16(
            false, afrag[fm].v, false, bfrag[fn].v,
            (short)0, acc[fm][fn], false, false);

    // my next-buffer writes landed; barrier makes them (and everyone's reads
    // of the current buffer) visible to all waves before the buffers swap.
    wait_async_zero();
    __syncthreads();
  }

  // epilogue: C += acc (16x16 f32 C layout: VGPR j -> M=j (lanes 0-15) / M=j+8 (lanes 16-31))
  const int colbase = blockN + wn * 32 + (lane & 15);
  const int rowadd  = (lane >> 4) * 8;
#pragma unroll
  for (int fm = 0; fm < 4; ++fm) {
#pragma unroll
    for (int fn = 0; fn < 2; ++fn) {
      const int col = colbase + fn * 16;
#pragma unroll
      for (int j = 0; j < 8; ++j) {
        const int row = blockM + wm * 64 + fm * 16 + rowadd + j;
        if (row < M) C[(size_t)row * N + col] += acc[fm][fn][j];
      }
    }
  }
}

// ---------------------------------------------------------------------------
// Host orchestration
// ---------------------------------------------------------------------------

static inline dim3 grid1d(long long n, int bs = 256) {
  return dim3((unsigned)((n + bs - 1) / bs));
}

extern "C" void kernel_launch(void* const* d_in, const int* in_sizes, int n_in,
                              void* d_out, int out_size, void* d_ws, size_t ws_size,
                              hipStream_t stream) {
  const float*     x       = (const float*)d_in[0];
  const float*     Wself1  = (const float*)d_in[1];
  const float*     Wneigh1 = (const float*)d_in[2];
  const float*     b1      = (const float*)d_in[3];
  const float*     Wself2  = (const float*)d_in[4];
  const float*     Wneigh2 = (const float*)d_in[5];
  const float*     b2      = (const float*)d_in[6];
  const long long* esrc    = (const long long*)d_in[7];   // [NREL][NE]
  const long long* edst    = (const long long*)d_in[8];   // [NREL][NE]
  float* out = (float*)d_out;                             // [NN][OUTF] fp32

  char* ws = (char*)d_ws;
  size_t off = 0;
  auto wsalloc = [&](size_t bytes) -> void* {
    void* p = ws + off;
    off = (off + bytes + 255) & ~(size_t)255;
    return p;
  };

  bf16_t*   xb   = (bf16_t*)  wsalloc((size_t)NN * INF  * 2);   // x in bf16
  bf16_t*   h1b  = (bf16_t*)  wsalloc((size_t)NN * HIDF * 2);   // relu(layer1) in bf16
  uint32_t* nb   = (uint32_t*)wsalloc((size_t)NN * F2   * 4);   // neighbor accum, packed bf16x2
  float*    out1 = (float*)   wsalloc((size_t)NN * HIDF * 4);   // layer1 fp32 accumulator
  float*    deg  = (float*)   wsalloc((size_t)NREL * NN * 4);
  bf16_t*   Ws1t = (bf16_t*)  wsalloc((size_t)HIDF * KDIM * 2);
  bf16_t*   Wn1t = (bf16_t*)  wsalloc((size_t)NREL * HIDF * KDIM * 2);
  bf16_t*   Ws2t = (bf16_t*)  wsalloc((size_t)OUTF * KDIM * 2);
  bf16_t*   Wn2t = (bf16_t*)  wsalloc((size_t)NREL * OUTF * KDIM * 2);
  float*    b1s  = (float*)   wsalloc(HIDF * 4);
  float*    b2s  = (float*)   wsalloc(OUTF * 4);

  // ---- weight / bias prep ----
  sum_self_transpose_kernel<<<grid1d((long long)KDIM * HIDF), 256, 0, stream>>>(Wself1, Ws1t, KDIM, HIDF);
  sum_self_transpose_kernel<<<grid1d((long long)KDIM * OUTF), 256, 0, stream>>>(Wself2, Ws2t, KDIM, OUTF);
  transpose_rel_kernel<<<grid1d((long long)NREL * KDIM * HIDF), 256, 0, stream>>>(Wneigh1, Wn1t, KDIM, HIDF);
  transpose_rel_kernel<<<grid1d((long long)NREL * KDIM * OUTF), 256, 0, stream>>>(Wneigh2, Wn2t, KDIM, OUTF);
  bias_sum_kernel<<<grid1d(HIDF), 256, 0, stream>>>(b1, b1s, HIDF);
  bias_sum_kernel<<<grid1d(OUTF), 256, 0, stream>>>(b2, b2s, OUTF);

  // ---- x -> bf16, degrees (shared by both layers) ----
  f32_to_bf16_kernel<<<grid1d((long long)NN * INF), 256, 0, stream>>>(x, xb, (long long)NN * INF);
  zero_u32_kernel<<<grid1d((long long)NREL * NN), 256, 0, stream>>>((uint32_t*)deg, (long long)NREL * NN);
  deg_kernel<<<grid1d((long long)NREL * NE), 256, 0, stream>>>(edst, deg);

  const dim3 blk(256);
  const dim3 g1(HIDF / NT, (NN + MT - 1) / MT);   // layer1 GEMM grid
  const dim3 g2(OUTF / NT, (NN + MT - 1) / MT);   // layer2 GEMM grid

  // ---- layer 1: out1 = b1s + x@Ws1sum + sum_r neigh_r@Wn1[r] ----
  init_c_bias_kernel<<<grid1d((long long)NN * HIDF), 256, 0, stream>>>(out1, b1s, (long long)NN * HIDF, HIDF);
  wmma_gemm_acc_kernel<<<g1, blk, 0, stream>>>(xb, Ws1t, out1, NN, HIDF);
  for (int r = 0; r < NREL; ++r) {
    zero_u32_kernel<<<grid1d((long long)NN * F2), 256, 0, stream>>>(nb, (long long)NN * F2);
    scatter_pk_bf16_kernel<<<dim3(NE), blk, 0, stream>>>(
        (const uint32_t*)xb, esrc + (size_t)r * NE, edst + (size_t)r * NE, nb);
    mean_div_kernel<<<dim3(NN), blk, 0, stream>>>(nb, deg + (size_t)r * NN);
    wmma_gemm_acc_kernel<<<g1, blk, 0, stream>>>(
        (const bf16_t*)nb, Wn1t + (size_t)r * HIDF * KDIM, out1, NN, HIDF);
  }
  relu_f32_to_bf16_kernel<<<grid1d((long long)NN * HIDF), 256, 0, stream>>>(out1, h1b, (long long)NN * HIDF);

  // ---- layer 2: out = b2s + h1@Ws2sum + sum_r neigh_r@Wn2[r] ----
  init_c_bias_kernel<<<grid1d((long long)NN * OUTF), 256, 0, stream>>>(out, b2s, (long long)NN * OUTF, OUTF);
  wmma_gemm_acc_kernel<<<g2, blk, 0, stream>>>(h1b, Ws2t, out, NN, OUTF);
  for (int r = 0; r < NREL; ++r) {
    zero_u32_kernel<<<grid1d((long long)NN * F2), 256, 0, stream>>>(nb, (long long)NN * F2);
    scatter_pk_bf16_kernel<<<dim3(NE), blk, 0, stream>>>(
        (const uint32_t*)h1b, esrc + (size_t)r * NE, edst + (size_t)r * NE, nb);
    mean_div_kernel<<<dim3(NN), blk, 0, stream>>>(nb, deg + (size_t)r * NN);
    wmma_gemm_acc_kernel<<<g2, blk, 0, stream>>>(
        (const bf16_t*)nb, Wn2t + (size_t)r * OUTF * KDIM, out, NN, OUTF);
  }
}
